// RNN_model_20916490731782
// MI455X (gfx1250) — matrix operand order repaired and checked by reading, not compiled, not run on hardware
//
#include <hip/hip_runtime.h>
#include <hip/hip_bf16.h>

#define B_ 32
#define T_ 512
#define D_ 2048
#define H_ 128
#define G_ 512

typedef __attribute__((ext_vector_type(16))) __bf16        v16bf;
typedef __attribute__((ext_vector_type(8)))  float         v8f;
typedef __attribute__((ext_vector_type(8)))  unsigned int  v8u;

__device__ __forceinline__ __bf16 f2bf(float f) { return (__bf16)f; }
__device__ __forceinline__ float  bf2f(__bf16 b) { return (float)b; }

// A-matrix 16x32 bf16 fragment: lanes 0-15 M=lane K=0..15,
// lanes 16-31 M=lane-16 K=16..31. K-pairs are contiguous -> 32-bit loads.
// s row-major [rows][ld]; ld must be even, k0 multiple of 32.
__device__ __forceinline__ v16bf load_a_frag(const __bf16* s, int row0, int ld,
                                             int k0, int lane) {
  const int m = lane & 15;
  const int half = (lane >> 4) & 1;
  const unsigned int* p =
      (const unsigned int*)(s + (row0 + m) * ld + k0 + half * 16);
  v8u u;
#pragma unroll
  for (int v = 0; v < 8; ++v) {
    const int kw = v;   // dword index = k/2
    u[v] = p[kw];
  }
  return __builtin_bit_cast(v16bf, u);
}

// B-matrix 32x16 bf16 fragment from LDS staging: lanes 0-15 N=lane K=0..15,
// lanes 16-31 K=16..31. s row-major [K][ld].
__device__ __forceinline__ v16bf load_b_frag(const __bf16* s, int k0, int ld,
                                             int n0, int lane) {
  const int n = lane & 15;
  const int half = (lane >> 4) & 1;
  const __bf16* p = s + (k0 + half * 16) * ld + n0 + n;
  v16bf b;
#pragma unroll
  for (int v = 0; v < 8; ++v) {
    b[2 * v]     = p[(2 * v) * ld];
    b[2 * v + 1] = p[(2 * v + 1) * ld];
  }
  return b;
}

__device__ __forceinline__ v8f wmma_bf16(v16bf a, v16bf b, v8f c) {
  return __builtin_amdgcn_wmma_f32_16x16x32_bf16(false, a, false, b, (short)0, c,
                                                 false, false);
}

__device__ __forceinline__ float sigm(float x) { return 1.0f / (1.0f + __expf(-x)); }
__device__ __forceinline__ float tanh_(float x) { return 2.0f / (1.0f + __expf(-2.0f * x)) - 1.0f; }

// ---------------------------------------------------------------------------
// Kernel 0: pre-swizzle W_ih_f [512,2048] f32 -> bf16 fragment-ready layout.
// Fragment f = kc*32 + nt holds 512 bf16: element (lane, i) = W[n][k],
// n = nt*16 + (lane&15), k = kc*32 + (lane>>4)*16 + i. Each lane's 16 values
// are contiguous (32 B) -> GEMM fetches a whole fragment with b128 loads.
// ---------------------------------------------------------------------------
__global__ __launch_bounds__(256) void k_prep_w(const float* __restrict__ W,
                                                __bf16* __restrict__ wf) {
  const int o0 = (blockIdx.x * 256 + threadIdx.x) * 4;   // 4 contiguous outputs
  const int f = o0 >> 9;
  const int r = o0 & 511;
  const int lane = r >> 4, i = r & 15;
  const int kc = f >> 5, nt = f & 31;
  const int n = nt * 16 + (lane & 15);
  const int k = kc * 32 + ((lane >> 4) << 4) + i;
  const float* src = W + (size_t)n * D_ + k;
#pragma unroll
  for (int j = 0; j < 4; ++j) wf[o0 + j] = f2bf(src[j]);
}

// ---------------------------------------------------------------------------
// Kernel 1: xp[t][b][g] = (x @ W_ih^T)[b*T+t][g] + b_ih[g] + b_hh[g]
// BM=64, BN=512 (full width -> x streamed from HBM exactly once).
// 256 threads / 8 waves; wave tile 32(M) x 128(N) = 16 C fragments.
// B fragments come pre-swizzled from global (L2-resident, 2 MB).
// ---------------------------------------------------------------------------
#define LDA1 34
__global__ __launch_bounds__(256) void k_gemm_xp(const float* __restrict__ x,
                                                 const __bf16* __restrict__ wf,
                                                 const float* __restrict__ bi,
                                                 const float* __restrict__ bh,
                                                 float* __restrict__ xp) {
  __shared__ __align__(16) __bf16 sA[64 * LDA1];
  const int tid = threadIdx.x;
  const int lane = tid & 31;
  const int wave = tid >> 5;
  const int wrow = wave >> 2;          // 0..1 -> 32-row strip
  const int wcol = wave & 3;           // 0..3 -> 128-col strip
  const int mblk = blockIdx.x * 64;

  v8f acc[2][8] = {};
  for (int kb = 0, kc = 0; kb < D_; kb += 32, ++kc) {
#pragma unroll
    for (int i = 0; i < 8; ++i) {      // A: 64x32 f32 -> bf16
      int e = tid + i * 256;
      int rr = e >> 5, cc = e & 31;
      sA[rr * LDA1 + cc] = f2bf(x[(size_t)(mblk + rr) * D_ + kb + cc]);
    }
    __syncthreads();
    v16bf a0 = load_a_frag(sA, wrow * 32,      LDA1, 0, lane);
    v16bf a1 = load_a_frag(sA, wrow * 32 + 16, LDA1, 0, lane);
#pragma unroll
    for (int nt = 0; nt < 8; ++nt) {
      int fr = kc * 32 + wcol * 8 + nt;
      const v8u* pb = (const v8u*)(wf + (size_t)fr * 512 + lane * 16);
      v16bf b = __builtin_bit_cast(v16bf, *pb);
      acc[0][nt] = wmma_bf16(a0, b, acc[0][nt]);
      acc[1][nt] = wmma_bf16(a1, b, acc[1][nt]);
    }
    __syncthreads();
  }
  const int halfsel = (lane >> 4) & 1;
#pragma unroll
  for (int mi = 0; mi < 2; ++mi) {
#pragma unroll
    for (int nt = 0; nt < 8; ++nt) {
      int col = wcol * 128 + nt * 16 + (lane & 15);
      float bias = bi[col] + bh[col];
#pragma unroll
      for (int v = 0; v < 8; ++v) {
        int row = mblk + wrow * 32 + mi * 16 + (halfsel ? v + 8 : v);
        int b = row >> 9;              // row / T
        int t = row & (T_ - 1);        // row % T
        xp[((size_t)t * B_ + b) * G_ + col] = acc[mi][nt][v] + bias;
      }
    }
  }
}

// ---------------------------------------------------------------------------
// Kernel 2: backward direction = ONE cell step at t=T-1 from zero state
// (reverse scan's output aligned to t=T-1 is its first step; W_hh_b * h0 = 0).
// One workgroup, 512 threads (16 waves).
// ---------------------------------------------------------------------------
#define LDB2 520
__global__ __launch_bounds__(512) void k_bwd_step(const float* __restrict__ x,
                                                  const float* __restrict__ W,
                                                  const float* __restrict__ bi,
                                                  const float* __restrict__ bh,
                                                  float* __restrict__ hb) {
  __shared__ __align__(16) __bf16 sA[32 * LDA1];
  __shared__ __align__(16) __bf16 sB[32 * LDB2];
  __shared__ float  sg[32 * G_];       // gates f32
  const int tid = threadIdx.x;
  const int lane = tid & 31;
  const int wave = tid >> 5;           // 0..15 -> 32-col strip
  const int n0 = wave * 32;

  v8f acc[2][2] = {};
  for (int kb = 0; kb < D_; kb += 32) {
#pragma unroll
    for (int i = 0; i < 2; ++i) {      // A: 32x32 = x[b][T-1][kb+c]
      int e = tid + i * 512;
      int r = e >> 5, c = e & 31;
      sA[r * LDA1 + c] = f2bf(x[((size_t)r * T_ + (T_ - 1)) * D_ + kb + c]);
    }
#pragma unroll
    for (int i = 0; i < 32; ++i) {     // B[k][n] = W[n][k]: 32x512
      int e = tid + i * 512;
      int n = e >> 5, k = e & 31;
      sB[k * LDB2 + n] = f2bf(W[(size_t)n * D_ + kb + k]);
    }
    __syncthreads();
    v16bf a0 = load_a_frag(sA, 0,  LDA1, 0, lane);
    v16bf a1 = load_a_frag(sA, 16, LDA1, 0, lane);
    v16bf b0 = load_b_frag(sB, 0, LDB2, n0,      lane);
    v16bf b1 = load_b_frag(sB, 0, LDB2, n0 + 16, lane);
    acc[0][0] = wmma_bf16(a0, b0, acc[0][0]);
    acc[0][1] = wmma_bf16(a0, b1, acc[0][1]);
    acc[1][0] = wmma_bf16(a1, b0, acc[1][0]);
    acc[1][1] = wmma_bf16(a1, b1, acc[1][1]);
    __syncthreads();
  }
  const int halfsel = (lane >> 4) & 1;
#pragma unroll
  for (int mi = 0; mi < 2; ++mi) {
#pragma unroll
    for (int ni = 0; ni < 2; ++ni) {
      int col = n0 + ni * 16 + (lane & 15);
      float bias = bi[col] + bh[col];
#pragma unroll
      for (int v = 0; v < 8; ++v) {
        int row = mi * 16 + (halfsel ? v + 8 : v);
        sg[row * G_ + col] = acc[mi][ni][v] + bias;
      }
    }
  }
  __syncthreads();
#pragma unroll
  for (int i = 0; i < 8; ++i) {        // cell update, c0 = 0 (f-gate irrelevant)
    int e = tid + i * 512;
    int b = e >> 7, j = e & 127;
    float ig = sg[b * G_ + j];
    float gg = sg[b * G_ + 256 + j];
    float og = sg[b * G_ + 384 + j];
    float c = sigm(ig) * tanh_(gg);
    hb[b * H_ + j] = sigm(og) * tanh_(c);
  }
}

// ---------------------------------------------------------------------------
// Kernel 3: forward recurrence, persistent single workgroup (8 waves).
// W_hh^T fragments resident in registers; c kept in accumulator fragments;
// h round-trips LDS as bf16 each step.
// ---------------------------------------------------------------------------
#define LDW3 520
#define LDH3 136
__global__ __launch_bounds__(256) void k_fwd_scan(const float* __restrict__ xp,
                                                  const float* __restrict__ Whh,
                                                  float* __restrict__ hf) {
  __shared__ __align__(16) __bf16 sW[H_ * LDW3];   // [k][n], n = gate col
  __shared__ __align__(16) __bf16 sh[B_ * LDH3];   // h state, bf16
  const int tid = threadIdx.x;
  const int lane = tid & 31;
  const int wave = tid >> 5;           // 0..7 -> hidden cols [wave*16, +16)
  const int j0 = wave * 16;

#pragma unroll 4
  for (int i = 0; i < 256; ++i) {      // sW[k][n] = Whh[n][k]
    int e = tid + i * 256;
    int k = e & 127, n = e >> 7;
    sW[k * LDW3 + n] = f2bf(Whh[(size_t)n * H_ + k]);
  }
#pragma unroll
  for (int i = 0; i < 17; ++i) {       // zero sh (incl. padding)
    int e = tid + i * 256;
    if (e < B_ * LDH3) sh[e] = f2bf(0.0f);
  }
  __syncthreads();

  v16bf Bf[4][4];                      // [gate][kchunk], resident in VGPRs
#pragma unroll
  for (int g = 0; g < 4; ++g)
#pragma unroll
    for (int kc = 0; kc < 4; ++kc)
      Bf[g][kc] = load_b_frag(sW, kc * 32, LDW3, g * 128 + j0, lane);

  const int halfsel = (lane >> 4) & 1;
  const int colLoc = j0 + (lane & 15);
  v8f cc[2] = {};                      // cell state fragments (mt = 0,1)

  for (int t = 0; t < T_; ++t) {
    v8f acc[2][4];
    const float* xpt = xp + (size_t)t * B_ * G_;
#pragma unroll
    for (int mt = 0; mt < 2; ++mt)
#pragma unroll
      for (int g = 0; g < 4; ++g)
#pragma unroll
        for (int v = 0; v < 8; ++v) {
          int row = mt * 16 + (halfsel ? v + 8 : v);
          acc[mt][g][v] = xpt[(size_t)row * G_ + g * 128 + colLoc];
        }
#pragma unroll
    for (int kc = 0; kc < 4; ++kc) {
      v16bf a0 = load_a_frag(sh, 0,  LDH3, kc * 32, lane);
      v16bf a1 = load_a_frag(sh, 16, LDH3, kc * 32, lane);
#pragma unroll
      for (int g = 0; g < 4; ++g) {
        acc[0][g] = wmma_bf16(a0, Bf[g][kc], acc[0][g]);
        acc[1][g] = wmma_bf16(a1, Bf[g][kc], acc[1][g]);
      }
    }
    __syncthreads();                   // all h reads done before overwrite
#pragma unroll
    for (int mt = 0; mt < 2; ++mt)
#pragma unroll
      for (int v = 0; v < 8; ++v) {
        float iv = sigm(acc[mt][0][v]);
        float fv = sigm(acc[mt][1][v]);
        float gv = tanh_(acc[mt][2][v]);
        float ov = sigm(acc[mt][3][v]);
        cc[mt][v] = fv * cc[mt][v] + iv * gv;
        float hv = ov * tanh_(cc[mt][v]);
        int row = mt * 16 + (halfsel ? v + 8 : v);
        sh[row * LDH3 + colLoc] = f2bf(hv);
      }
    __syncthreads();                   // h visible before next step's reads
  }
#pragma unroll
  for (int i = 0; i < 16; ++i) {
    int e = tid + i * 256;
    int b = e >> 7, j = e & 127;
    hf[e] = bf2f(sh[b * LDH3 + j]);
  }
}

// ---------------------------------------------------------------------------
// Kernel 4: MLP head + softmax (tiny, plain f32 VALU). One block, 256 threads.
// ---------------------------------------------------------------------------
__global__ __launch_bounds__(256) void k_head(const float* __restrict__ hf,
                                              const float* __restrict__ hb,
                                              const float* __restrict__ W1,
                                              const float* __restrict__ b1,
                                              const float* __restrict__ W2,
                                              const float* __restrict__ b2,
                                              const float* __restrict__ W3,
                                              const float* __restrict__ b3,
                                              float* __restrict__ out) {
  __shared__ float last[32 * 256];
  __shared__ float o1[32 * 256];
  __shared__ float o2[32 * 64];
  __shared__ float o3[32 * 11];
  const int tid = threadIdx.x;
#pragma unroll
  for (int i = 0; i < 32; ++i) {
    int e = tid + i * 256;
    int b = e >> 8, j = e & 255;
    last[e] = (j < 128) ? hf[b * 128 + j] : hb[b * 128 + (j - 128)];
  }
  __syncthreads();
  for (int i = 0; i < 32; ++i) {       // fc1: [32,256] x [256,256]^T
    int e = tid + i * 256;
    int b = e >> 8, j = e & 255;
    float s = b1[j];
    for (int k = 0; k < 256; ++k) s += last[b * 256 + k] * W1[j * 256 + k];
    o1[e] = s;
  }
  __syncthreads();
  for (int i = 0; i < 8; ++i) {        // fc2: [32,256] x [64,256]^T
    int e = tid + i * 256;
    int b = e >> 6, j = e & 63;
    float s = b2[j];
    for (int k = 0; k < 256; ++k) s += o1[b * 256 + k] * W2[j * 256 + k];
    o2[e] = s;
  }
  __syncthreads();
  if (tid < 352) {                     // fc3: [32,64] x [11,64]^T
    int b = tid / 11, j = tid % 11;
    float s = b3[j];
    for (int k = 0; k < 64; ++k) s += o2[b * 64 + k] * W3[j * 64 + k];
    o3[b * 11 + j] = s;
  }
  __syncthreads();
  if (tid < 32) {                      // softmax over 11 classes
    float m = -1e30f;
    for (int j = 0; j < 11; ++j) m = fmaxf(m, o3[tid * 11 + j]);
    float e[11], sum = 0.0f;
    for (int j = 0; j < 11; ++j) { e[j] = __expf(o3[tid * 11 + j] - m); sum += e[j]; }
    float inv = 1.0f / sum;
    for (int j = 0; j < 11; ++j) out[tid * 11 + j] = e[j] * inv;
  }
}

extern "C" void kernel_launch(void* const* d_in, const int* in_sizes, int n_in,
                              void* d_out, int out_size, void* d_ws, size_t ws_size,
                              hipStream_t stream) {
  (void)in_sizes; (void)n_in; (void)out_size; (void)ws_size;
  const float* x    = (const float*)d_in[0];
  const float* Wihf = (const float*)d_in[1];
  const float* Whhf = (const float*)d_in[2];
  const float* bihf = (const float*)d_in[3];
  const float* bhhf = (const float*)d_in[4];
  const float* Wihb = (const float*)d_in[5];
  // d_in[6] = W_hh_b: multiplies h0 == 0 in the only backward step that
  // reaches the output -> mathematically unused.
  const float* bihb = (const float*)d_in[7];
  const float* bhhb = (const float*)d_in[8];
  const float* W1 = (const float*)d_in[9];
  const float* b1 = (const float*)d_in[10];
  const float* W2 = (const float*)d_in[11];
  const float* b2 = (const float*)d_in[12];
  const float* W3 = (const float*)d_in[13];
  const float* b3 = (const float*)d_in[14];
  float* out = (float*)d_out;

  char* ws = (char*)d_ws;
  float* xp = (float*)ws;                                   // [T][B][G] f32, 32 MB
  float*  hf = (float*)(ws + (size_t)T_ * B_ * G_ * sizeof(float));  // [32][128]
  float*  hb = hf + B_ * H_;                                // [32][128]
  __bf16* wf = (__bf16*)(ws + (size_t)T_ * B_ * G_ * sizeof(float)
                            + 2 * B_ * H_ * sizeof(float)); // 2 MB, 1KB-aligned

  k_prep_w<<<1024, 256, 0, stream>>>(Wihf, wf);
  k_gemm_xp<<<256, 256, 0, stream>>>(x, wf, bihf, bhhf, xp);
  k_bwd_step<<<1, 512, 0, stream>>>(x, Wihb, bihb, bhhb, hb);
  k_fwd_scan<<<1, 256, 0, stream>>>(xp, Whhf, hf);
  k_head<<<1, 256, 0, stream>>>(hf, hb, W1, b1, W2, b2, W3, b3, out);
}
